// LocalWindowAttention_13924283973718
// MI455X (gfx1250) — matrix operand order repaired and checked by reading, not compiled, not run on hardware
//
#include <hip/hip_runtime.h>
#include <hip/hip_bf16.h>

typedef __attribute__((ext_vector_type(16))) __bf16 v16bf;
typedef __attribute__((ext_vector_type(8)))  __bf16 v8bf;
typedef __attribute__((ext_vector_type(8)))  float  v8f;

#define WMMA_BF16(a, b, c) \
  __builtin_amdgcn_wmma_f32_16x16x32_bf16(false, (a), false, (b), (short)0, (c), false, false)

union FragU { v16bf v; v8bf h[2]; };

static __device__ inline v16bf make_frag(const __bf16* p0, const __bf16* p1) {
  FragU u;
  u.h[0] = *(const v8bf*)p0;
  u.h[1] = *(const v8bf*)p1;
  return u.v;
}

static __device__ inline v8f v8f_zero() {
  v8f z;
  #pragma unroll
  for (int i = 0; i < 8; ++i) z[i] = 0.0f;
  return z;
}

// ---------------------------------------------------------------------------
// Tiled GEMM with double-buffered LDS:  C[M x N] = A[M x K] * W[K x N] + bias
// Block tile 256x64, K-step 32. 256 threads = 8 waves.
// Wave w computes rows [w*32, w*32+32) x all 64 cols = 2x4 WMMA tiles/K-step
// (8 WMMA per 12 fragment ds_loads; half the barriers per FLOP vs 2x2).
// One barrier per K-step: iter i reads lds[cur], writes lds[cur^1]; reads of
// lds[cur^1] from iter i-1 finished before iter i-1's barrier.
// ---------------------------------------------------------------------------
template <bool A_F32, bool OUT_F32>
__global__ __launch_bounds__(256) void gemm_bias_kernel(
    const void* __restrict__ Ap, const float* __restrict__ W,
    const float* __restrict__ bias, void* __restrict__ Cp,
    int M, int N, int K) {
  constexpr int SA = 40;  // padded LDS row stride (halves), multiple of 8
  constexpr int SB = 40;
  __shared__ alignas(16) __bf16 lA[2][256 * SA];
  __shared__ alignas(16) __bf16 lB[2][64 * SB];   // transposed: [n][k]

  const int t    = threadIdx.x;
  const int lane = t & 31;
  const int w    = t >> 5;
  const int hi   = (lane >> 4) & 1;
  const int ln   = lane & 15;
  const int m0   = blockIdx.x * 256;
  const int n0   = blockIdx.y * 64;

  v8f acc[2][4];
  #pragma unroll
  for (int i = 0; i < 2; ++i)
    #pragma unroll
    for (int j = 0; j < 4; ++j) acc[i][j] = v8f_zero();

  const int arow = t;               // one 32-wide A row slice per thread
  const int bk   = t >> 3;          // 0..31
  const int bnb  = (t & 7) * 8;     // 0..56

  v8bf ra[4];          // staged A (bf16), 32 halves
  float4 rb0, rb1;     // staged W (fp32)

  auto gload = [&](int k0) {
    if constexpr (A_F32) {
      const float* A = (const float*)Ap + (size_t)(m0 + arow) * K + k0;
      #pragma unroll
      for (int q = 0; q < 4; ++q) {
        float4 f0 = ((const float4*)A)[2 * q];
        float4 f1 = ((const float4*)A)[2 * q + 1];
        ra[q] = v8bf{(__bf16)f0.x, (__bf16)f0.y, (__bf16)f0.z, (__bf16)f0.w,
                     (__bf16)f1.x, (__bf16)f1.y, (__bf16)f1.z, (__bf16)f1.w};
      }
    } else {
      const __bf16* A = (const __bf16*)Ap + (size_t)(m0 + arow) * K + k0;
      #pragma unroll
      for (int q = 0; q < 4; ++q) ra[q] = ((const v8bf*)A)[q];
    }
    const float* Wp = W + (size_t)(k0 + bk) * N + n0 + bnb;
    rb0 = ((const float4*)Wp)[0];
    rb1 = ((const float4*)Wp)[1];
  };
  auto sstore = [&](int buf) {
    #pragma unroll
    for (int q = 0; q < 4; ++q)
      *(v8bf*)&lA[buf][arow * SA + 8 * q] = ra[q];
    lB[buf][(bnb + 0) * SB + bk] = (__bf16)rb0.x;
    lB[buf][(bnb + 1) * SB + bk] = (__bf16)rb0.y;
    lB[buf][(bnb + 2) * SB + bk] = (__bf16)rb0.z;
    lB[buf][(bnb + 3) * SB + bk] = (__bf16)rb0.w;
    lB[buf][(bnb + 4) * SB + bk] = (__bf16)rb1.x;
    lB[buf][(bnb + 5) * SB + bk] = (__bf16)rb1.y;
    lB[buf][(bnb + 6) * SB + bk] = (__bf16)rb1.z;
    lB[buf][(bnb + 7) * SB + bk] = (__bf16)rb1.w;
  };

  gload(0);
  sstore(0);
  __syncthreads();

  const int ksteps = K >> 5;
  int cur = 0;
  for (int ks = 0; ks < ksteps; ++ks) {
    const bool has_next = (ks + 1) < ksteps;
    if (has_next) gload((ks + 1) << 5);                 // overlap with compute
    if (ks + 2 < ksteps) {                              // L2 prefetch 2 ahead
      const size_t esz = A_F32 ? 4 : 2;
      __builtin_prefetch((const char*)Ap +
                         ((size_t)(m0 + arow) * K + ((ks + 2) << 5)) * esz, 0, 0);
    }

    v16bf af[2], bfg[4];
    #pragma unroll
    for (int i = 0; i < 2; ++i) {
      const __bf16* base = &lA[cur][(w * 32 + i * 16 + ln) * SA];
      af[i] = make_frag(base + hi * 16, base + hi * 16 + 8);
    }
    #pragma unroll
    for (int j = 0; j < 4; ++j) {
      const __bf16* base = &lB[cur][(j * 16 + ln) * SB + hi * 16];
      bfg[j] = make_frag(base, base + 8);
    }
    #pragma unroll
    for (int i = 0; i < 2; ++i)
      #pragma unroll
      for (int j = 0; j < 4; ++j)
        acc[i][j] = WMMA_BF16(af[i], bfg[j], acc[i][j]);

    if (has_next) {
      sstore(cur ^ 1);
      __syncthreads();
    }
    cur ^= 1;
  }

  // ---- epilogue: bias add + store ----
  #pragma unroll
  for (int j = 0; j < 4; ++j) {
    const int col = n0 + j * 16 + ln;
    const float bv = bias[col];
    #pragma unroll
    for (int i = 0; i < 2; ++i) {
      const int rbase = m0 + w * 32 + i * 16 + hi * 8;
      #pragma unroll
      for (int vg = 0; vg < 8; ++vg) {
        const float val = acc[i][j][vg] + bv;
        if constexpr (OUT_F32)
          ((float*)Cp)[(size_t)(rbase + vg) * N + col] = val;
        else
          ((__bf16*)Cp)[(size_t)(rbase + vg) * N + col] = (__bf16)val;
      }
    }
  }
}

// ---------------------------------------------------------------------------
// Banded attention, fully static band: one block per (batch, 64-query slab).
// Key band is ALWAYS 192 keys [q0-64, q0+128); out-of-range key rows are
// clamped on load (finite data) and masked to 0 probability in softmax, so
// they contribute exactly zero to P@V. No wave-uniform branch ladders.
// ---------------------------------------------------------------------------
__global__ __launch_bounds__(128) void attn_kernel(
    const __bf16* __restrict__ qkv, __bf16* __restrict__ att) {
  constexpr int S = 4096, E = 512, WIN = 64;
  constexpr int NKT = 12, NKC = 6;                     // 192 keys static
  constexpr int SQ = 72, SK = 72, SV = 200, SP = 200;  // padded strides

  __shared__ alignas(16) __bf16 sA[64 * SQ + 192 * SK];  // lq+lk, reused as lv
  __shared__ alignas(16) __bf16 lp[64 * SP];             // probs, bf16
  __bf16* lq = sA;
  __bf16* lk = sA + 64 * SQ;
  __bf16* lv = sA;  // aliases lq/lk after scores (barrier-separated)

  const int t    = threadIdx.x;
  const int lane = t & 31;
  const int w    = t >> 5;
  const int hi   = (lane >> 4) & 1;
  const int ln   = lane & 15;

  const int b      = blockIdx.x >> 6;
  const int qb     = blockIdx.x & 63;
  const int q0     = qb * 64;
  const int kstart = q0 - WIN;  // may be negative; may run past S

  const __bf16* qkv_b = qkv + (size_t)b * S * (3 * E);
  const __bf16* qsrc  = qkv_b + (size_t)q0 * (3 * E);

  v8f sacc[NKT];
  #pragma unroll
  for (int i = 0; i < NKT; ++i) sacc[i] = v8f_zero();

  // ================= scores: S = Q K^T over E in 64-wide chunks =============
  const int lrow = t >> 1;
  const int lcb  = (t & 1) * 32;
  int krow[3];
  #pragma unroll
  for (int rr = 0; rr < 3; ++rr) {
    int g = kstart + lrow + rr * 64;
    g = g < 0 ? 0 : (g > S - 1 ? S - 1 : g);   // clamp: masked later
    krow[rr] = g;
  }

  for (int ec = 0; ec < 8; ++ec) {
    const int e0 = ec * 64;
    {  // Q slab 64x64
      const uint4* src = (const uint4*)(qsrc + (size_t)lrow * (3 * E) + e0 + lcb);
      uint4* dst = (uint4*)&lq[lrow * SQ + lcb];
      dst[0] = src[0]; dst[1] = src[1]; dst[2] = src[2]; dst[3] = src[3];
    }
    #pragma unroll
    for (int rr = 0; rr < 3; ++rr) {  // K slab 192x64, rows clamped
      const uint4* src =
          (const uint4*)(qkv_b + (size_t)krow[rr] * (3 * E) + E + e0 + lcb);
      uint4* dst = (uint4*)&lk[(lrow + rr * 64) * SK + lcb];
      dst[0] = src[0]; dst[1] = src[1]; dst[2] = src[2]; dst[3] = src[3];
    }
    __syncthreads();

    v16bf aq0, aq1;
    {
      const __bf16* base = &lq[(w * 16 + ln) * SQ];
      aq0 = make_frag(base + hi * 16,      base + hi * 16 + 8);
      aq1 = make_frag(base + 32 + hi * 16, base + 40 + hi * 16);
    }
    #pragma unroll
    for (int kt = 0; kt < NKT; ++kt) {
      const __bf16* kb = &lk[(kt * 16 + ln) * SK + hi * 16];
      v16bf b0 = make_frag(kb,      kb + 8);
      v16bf b1 = make_frag(kb + 32, kb + 40);
      sacc[kt] = WMMA_BF16(aq0, b0, sacc[kt]);
      sacc[kt] = WMMA_BF16(aq1, b1, sacc[kt]);
    }
    __syncthreads();
  }

  // ================= masked softmax (fp32) =================================
  const float scale = 0.04419417382415922f;  // 1/sqrt(512)
  #pragma unroll
  for (int i = 0; i < 8; ++i) {
    const int q = q0 + w * 16 + i + hi * 8;
    float vals[NKT];
    float mx = -3.0e38f;
    #pragma unroll
    for (int kt = 0; kt < NKT; ++kt) {
      const int key = kstart + kt * 16 + ln;
      const int d = q - key;
      const bool bad = ((unsigned)key >= (unsigned)S) | (d > WIN) | (d < -WIN);
      float s = bad ? -3.0e38f : sacc[kt][i] * scale;
      vals[kt] = s;
      mx = fmaxf(mx, s);
    }
    #pragma unroll
    for (int m = 8; m >= 1; m >>= 1) mx = fmaxf(mx, __shfl_xor(mx, m, 32));
    float sum = 0.0f;
    #pragma unroll
    for (int kt = 0; kt < NKT; ++kt) {
      const float p = __expf(vals[kt] - mx);  // masked -> 0
      vals[kt] = p;
      sum += p;
    }
    #pragma unroll
    for (int m = 8; m >= 1; m >>= 1) sum += __shfl_xor(sum, m, 32);
    const float inv = 1.0f / sum;
    const int prow = (w * 16 + i + hi * 8) * SP;
    #pragma unroll
    for (int kt = 0; kt < NKT; ++kt)
      lp[prow + kt * 16 + ln] = (__bf16)(vals[kt] * inv);
  }
  __syncthreads();

  // ================= attended = P @ V over E in 64-wide chunks ==============
  v16bf pf[NKC];
  {
    const __bf16* base = &lp[(w * 16 + ln) * SP];
    #pragma unroll
    for (int c = 0; c < NKC; ++c)
      pf[c] = make_frag(base + c * 32 + hi * 16, base + c * 32 + hi * 16 + 8);
  }

  int vkey[12];  // clamped V rows for this thread's 12 (key, e-block) pairs
  #pragma unroll
  for (int it = 0; it < 12; ++it) {
    const int p = t + it * 128;
    int g = kstart + (p >> 3);
    g = g < 0 ? 0 : (g > S - 1 ? S - 1 : g);
    vkey[it] = g;
  }

  for (int ec = 0; ec < 8; ++ec) {
    const int e0 = ec * 64;
    #pragma unroll
    for (int it = 0; it < 12; ++it) {  // stage V transposed: lv[e][key]
      const int p  = t + it * 128;
      const int eb = (p & 7) * 8;
      const v8bf v =
          *(const v8bf*)(qkv_b + (size_t)vkey[it] * (3 * E) + 2 * E + e0 + eb);
      #pragma unroll
      for (int u = 0; u < 8; ++u) lv[(eb + u) * SV + (p >> 3)] = v[u];
    }
    __syncthreads();

    #pragma unroll
    for (int nt = 0; nt < 4; ++nt) {
      v8f o = v8f_zero();
      const __bf16* vb = &lv[(nt * 16 + ln) * SV];
      #pragma unroll
      for (int c = 0; c < NKC; ++c) {
        v16bf bfrag = make_frag(vb + c * 32 + hi * 16, vb + c * 32 + hi * 16 + 8);
        o = WMMA_BF16(pf[c], bfrag, o);
      }
      const int col = e0 + nt * 16 + ln;
      #pragma unroll
      for (int vg = 0; vg < 8; ++vg) {
        const int q = q0 + w * 16 + vg + hi * 8;
        att[((size_t)b * S + q) * E + col] = (__bf16)o[vg];
      }
    }
    __syncthreads();
  }
}

// ---------------------------------------------------------------------------
extern "C" void kernel_launch(void* const* d_in, const int* in_sizes, int n_in,
                              void* d_out, int out_size, void* d_ws, size_t ws_size,
                              hipStream_t stream) {
  const float* x     = (const float*)d_in[0];
  const float* W_qkv = (const float*)d_in[1];
  const float* b_qkv = (const float*)d_in[2];
  const float* W_out = (const float*)d_in[3];
  const float* b_out = (const float*)d_in[4];
  float* out = (float*)d_out;

  constexpr int Bb = 4, S = 4096, E = 512;
  constexpr int M = Bb * S;  // 16384

  __bf16* qkv = (__bf16*)d_ws;                                   // [M][3E] bf16
  __bf16* att = (__bf16*)((char*)d_ws + (size_t)M * 3 * E * 2);  // [M][E]  bf16

  dim3 g1(M / 256, (3 * E) / 64);
  gemm_bias_kernel<true, false><<<g1, 256, 0, stream>>>(x, W_qkv, b_qkv, qkv, M, 3 * E, E);

  attn_kernel<<<dim3(Bb * (S / 64)), 128, 0, stream>>>(qkv, att);

  dim3 g3(M / 256, E / 64);
  gemm_bias_kernel<false, true><<<g3, 256, 0, stream>>>(att, W_out, b_out, out, M, E, E);
}
